// MultiHeadGraphAttNet_46299747451283
// MI455X (gfx1250) — matrix operand order repaired and checked by reading, not compiled, not run on hardware
//
#include <hip/hip_runtime.h>
#include <hip/hip_bf16.h>

typedef _Float16 half_t;
typedef __attribute__((ext_vector_type(16))) _Float16 v16h;
typedef __attribute__((ext_vector_type(8)))  float    v8f;

#define DIM   256
#define HEADS 8
#define BS    8
#define NN    1024
#define NEGV  -1.0e9f

// workspace layout (bytes)
#define OFF_FEATS_H 0UL                                        // BS*NN*DIM f16   (4 MB)
#define OFF_FCW_T   (OFF_FEATS_H + (size_t)BS*NN*DIM*2)        // [col][k] f16    (1 MB)
#define OFF_FPW_T   (OFF_FCW_T   + (size_t)2048*DIM*2)         // [j][K]  f16     (1 MB)
#define OFF_FPT     (OFF_FPW_T   + (size_t)DIM*2048*2)         // [b][h][d][n] f16 (32 MB)
#define OFF_Q       (OFF_FPT     + (size_t)BS*HEADS*DIM*NN*2)  // f32 (256 KB)
#define OFF_K       (OFF_Q       + (size_t)BS*HEADS*NN*4)      // f32 (256 KB)
#define OFF_OUTH    (OFF_K       + (size_t)BS*HEADS*NN*4)      // [b*n][2048] f16 (32 MB)

// ---- gfx1250 async global->LDS path (guarded; falls back to sync copies) ----
// Builtin signature (from hipcc diagnostic): b128 takes (int4 AS1*, int4 AS3*, Ii, Ii),
// b32 takes (int AS1*, int AS3*, Ii, Ii). AS1 prints as "__device__" in HIP diags.
typedef int i32x4 __attribute__((vector_size(16)));
typedef __attribute__((address_space(1))) i32x4 g_i32x4;
typedef __attribute__((address_space(3))) i32x4 s_i32x4;
typedef __attribute__((address_space(1))) int   g_i32;
typedef __attribute__((address_space(3))) int   s_i32;

#if __has_builtin(__builtin_amdgcn_global_load_async_to_lds_b128)
#define ASYNC_B128(dst, src) \
  __builtin_amdgcn_global_load_async_to_lds_b128((g_i32x4*)(src), (s_i32x4*)(dst), 0, 0)
#define HAVE_ASYNC_LDS 1
#else
#define ASYNC_B128(dst, src) (*(float4*)(dst) = *(const float4*)(src))
#endif

#if __has_builtin(__builtin_amdgcn_global_load_async_to_lds_b32)
#define ASYNC_B32(dst, src) \
  __builtin_amdgcn_global_load_async_to_lds_b32((g_i32*)(src), (s_i32*)(dst), 0, 0)
#else
#define ASYNC_B32(dst, src) (*(float*)(dst) = *(const float*)(src))
#endif

#ifdef HAVE_ASYNC_LDS
#if __has_builtin(__builtin_amdgcn_s_wait_asynccnt)
#define ASYNC_WAIT() __builtin_amdgcn_s_wait_asynccnt(0)
#else
#define ASYNC_WAIT() asm volatile("s_wait_asynccnt 0" ::: "memory")
#endif
#else
#define ASYNC_WAIT() ((void)0)
#endif

__device__ inline v16h load16(const half_t* p) {
  v16h r;
  const float4* q = (const float4*)p;
  ((float4*)&r)[0] = q[0];
  ((float4*)&r)[1] = q[2];
  return r;
}

// ---------------- kernel 0: f32 -> f16 casts + weight transposes ----------------
__global__ __launch_bounds__(256) void k_convert(const float* feats, const float* fc_w,
                                                 const float* fp_w, half_t* feats_h,
                                                 half_t* fcw_t, half_t* fpw_t) {
  int idx = blockIdx.x * 256 + threadIdx.x;
  const int FE  = BS * NN * DIM;   // 2097152
  const int FCW = 2048 * DIM;      // 524288
  if (idx < FE) {
    feats_h[idx] = (half_t)feats[idx];
  } else if (idx < FE + FCW) {
    int i = idx - FE;
    int col = i >> 8, kk = i & 255;
    fcw_t[i] = (half_t)fc_w[(size_t)kk * 2048 + col];     // fc_w[k][col] -> [col][k]
  } else {
    int i = idx - FE - FCW;
    int j = i >> 11, kk = i & 2047;
    fpw_t[i] = (half_t)fp_w[(size_t)kk * 256 + j];        // fp_w[K][j] -> [j][K]
  }
}

// ---------------- kernel 1: feat_proj = feats @ fc_w + fc_b (stored transposed) ----
__global__ __launch_bounds__(256) void k_proj(const half_t* feats_h, const half_t* fcw_t,
                                              const float* fc_b, half_t* fpT) {
  __shared__ __align__(16) half_t As[16 * 264];
  int tid = threadIdx.x;
  int gid = blockIdx.x;
  int h  = gid & 7;
  int nt = (gid >> 3) & 63;
  int b  = gid >> 9;
  int n0 = nt * 16;
  {  // stage 16x256 A tile: async global->LDS, 32B per thread
    int r = tid >> 4, seg = tid & 15;
    const half_t* src = feats_h + ((size_t)(b * NN + n0 + r)) * DIM + seg * 16;
    half_t* dst = As + r * 264 + seg * 16;
    ASYNC_B128(dst, src);
    ASYNC_B128(dst + 8, src + 8);
  }
  ASYNC_WAIT();
  __syncthreads();
  int w = tid >> 5, lane = tid & 31;
  int row = lane & 15, hi = lane >> 4;
  v8f zero = {};
  v8f acc[2]; acc[0] = zero; acc[1] = zero;
  const half_t* Bbase = fcw_t + ((size_t)(h * 256 + w * 32 + row)) * DIM;
  #pragma unroll
  for (int k0 = 0; k0 < DIM; k0 += 32) {
    v16h af;
    const half_t* ap = As + row * 264 + k0 + hi * 8;
    ((float4*)&af)[0] = *(const float4*)ap;
    ((float4*)&af)[1] = *(const float4*)(ap + 16);
    #pragma unroll
    for (int t = 0; t < 2; ++t) {
      v16h bf = load16(Bbase + (size_t)t * 16 * DIM + k0 + hi * 8);
      acc[t] = __builtin_amdgcn_wmma_f32_16x16x32_f16(false, af, false, bf,
                                                      (short)0, acc[t], false, false);
    }
  }
  #pragma unroll
  for (int t = 0; t < 2; ++t) {
    int d = w * 32 + t * 16 + row;                 // channel within head
    float bias = fc_b[h * 256 + d];
    half_t* outp = fpT + ((size_t)((b * HEADS + h) * DIM + d)) * NN + n0;
    #pragma unroll
    for (int i = 0; i < 8; ++i) {
      int M = i + hi * 8;
      outp[M] = (half_t)(acc[t][i] + bias);
    }
  }
}

// ---------------- kernel 2: q/k scalar projections ------------------------------
__global__ __launch_bounds__(256) void k_qk(const half_t* fpT, const float* q_w,
                                            const float* q_b, const float* k_w,
                                            const float* k_b, float* Q, float* Kv) {
  int bh = blockIdx.x >> 2;
  int n  = ((blockIdx.x & 3) << 8) + threadIdx.x;
  const half_t* base = fpT + (size_t)bh * DIM * NN + n;
  float qa = 0.f, ka = 0.f;
  #pragma unroll 4
  for (int d = 0; d < DIM; ++d) {
    float x = (float)base[(size_t)d * NN];         // coalesced across lanes
    qa += x * q_w[d];
    ka += x * k_w[d];
  }
  Q[(size_t)bh * NN + n]  = qa + q_b[0];
  Kv[(size_t)bh * NN + n] = ka + k_b[0];
}

// ---------------- kernel 3: fused masked softmax + attn @ V ---------------------
__global__ __launch_bounds__(256) void k_attn(const float* adj, const half_t* fpT,
                                              const float* Q, const float* Kv,
                                              half_t* outh) {
  __shared__ unsigned char adjm[16 * 1025];
  __shared__ float ksl[HEADS * NN];
  __shared__ float rmaxs[HEADS * 16];
  __shared__ float rinvs[HEADS * 16];
  int tid = threadIdx.x;
  int nt = blockIdx.x & 63;
  int b  = blockIdx.x >> 6;
  int n0 = nt * 16;
  // k vectors for all heads: async copy into LDS (overlaps with adjm conversion)
  for (int idx = tid; idx < HEADS * NN; idx += 256)
    ASYNC_B32(&ksl[idx], &Kv[(size_t)b * HEADS * NN + idx]);
  // adjacency tile, shared by all 8 heads (needs VALU convert -> sync path)
  for (int idx = tid; idx < 16 * NN; idx += 256) {
    int r = idx >> 10, m = idx & 1023;
    adjm[r * 1025 + m] = (adj[((size_t)(b * NN + n0 + r)) * NN + m] != 0.0f) ? 1 : 0;
  }
  ASYNC_WAIT();
  __syncthreads();

  int h = tid >> 5, lane = tid & 31;                   // one wave per head
  const float* Qh = Q + ((size_t)(b * HEADS + h)) * NN + n0;
  // ---- pass 1: per-row max & sum of exp (online) ----
  for (int r = 0; r < 16; ++r) {
    float qv = Qh[r];
    float mx = -3.0e38f, sm = 0.0f;
    for (int m = lane; m < NN; m += 32) {
      float s = qv + ksl[h * NN + m] + (adjm[r * 1025 + m] ? 0.0f : NEGV);
      s = s > 0.0f ? s : 0.01f * s;                    // leaky_relu
      float nm = fmaxf(mx, s);
      sm = sm * __expf(mx - nm) + __expf(s - nm);
      mx = nm;
    }
    #pragma unroll
    for (int off = 16; off > 0; off >>= 1) {
      float mx2 = __shfl_xor(mx, off);
      float sm2 = __shfl_xor(sm, off);
      float nm = fmaxf(mx, mx2);
      sm = sm * __expf(mx - nm) + sm2 * __expf(mx2 - nm);
      mx = nm;
    }
    if (lane == 0) { rmaxs[h * 16 + r] = mx; rinvs[h * 16 + r] = 1.0f / sm; }
  }
  __syncthreads();

  // ---- pass 2: regenerate normalized probs as A-fragments, WMMA against V ----
  int row = lane & 15, hi = lane >> 4;
  float qv   = Qh[row];
  float rmax = rmaxs[h * 16 + row];
  float rinv = rinvs[h * 16 + row];
  const half_t* Vh = fpT + ((size_t)((b * HEADS + h)) * DIM) * NN;
  v8f zero = {};
  v8f acc[16];
  #pragma unroll
  for (int t = 0; t < 16; ++t) acc[t] = zero;
  for (int m0 = 0; m0 < NN; m0 += 32) {
    int kb = m0 + hi * 8;
    v16h af;
    #pragma unroll
    for (int e = 0; e < 16; ++e) {
      int m = kb + (e < 8 ? e : e + 8);                // ISA A-layout K interleave
      float s = qv + ksl[h * NN + m] + (adjm[row * 1025 + m] ? 0.0f : NEGV);
      s = s > 0.0f ? s : 0.01f * s;
      af[e] = (half_t)(__expf(s - rmax) * rinv);
    }
    #pragma unroll
    for (int t = 0; t < 16; ++t) {
      v16h bf = load16(Vh + ((size_t)(t * 16 + row)) * NN + m0 + hi * 8);
      acc[t] = __builtin_amdgcn_wmma_f32_16x16x32_f16(false, af, false, bf,
                                                      (short)0, acc[t], false, false);
    }
  }
  #pragma unroll
  for (int t = 0; t < 16; ++t) {
    int d = t * 16 + row;
    #pragma unroll
    for (int i = 0; i < 8; ++i) {
      int M = i + hi * 8;
      outh[((size_t)(b * NN + n0 + M)) * 2048 + h * 256 + d] = (half_t)acc[t][i];
    }
  }
}

// ---------------- kernel 4: out @ fp_w + fp_b + residual, sigmoid ---------------
__device__ inline void stage_final(half_t* buf, const half_t* outh, int b, int n0,
                                   int kc, int tid) {
  int r = tid >> 4, seg = tid & 15;
  const half_t* src = outh + ((size_t)(b * NN + n0 + r)) * 2048 + kc * 256 + seg * 16;
  half_t* dst = buf + r * 264 + seg * 16;
  ASYNC_B128(dst, src);
  ASYNC_B128(dst + 8, src + 8);
}

__global__ __launch_bounds__(256) void k_final(const half_t* outh, const half_t* fpw_t,
                                               const float* fp_b, const float* feats,
                                               float* out) {
  __shared__ __align__(16) half_t As[2][16 * 264];   // double-buffered K chunks
  int tid = threadIdx.x;
  int nt = blockIdx.x & 63;
  int b  = blockIdx.x >> 6;
  int n0 = nt * 16;
  int w = tid >> 5, lane = tid & 31;
  int row = lane & 15, hi = lane >> 4;
  v8f zero = {};
  v8f acc[2]; acc[0] = zero; acc[1] = zero;
  int jbase = w * 32 + row;
  stage_final(As[0], outh, b, n0, 0, tid);           // prologue fill
  #pragma unroll 1
  for (int kc = 0; kc < 8; ++kc) {                   // K = 2048 in 256-chunks
    int cur = kc & 1;
    ASYNC_WAIT();
    __syncthreads();                                 // chunk kc visible to all waves
    if (kc < 7) stage_final(As[1 - cur], outh, b, n0, kc + 1, tid);  // prefetch next
    const half_t* Ab = As[cur];
    #pragma unroll
    for (int k0 = 0; k0 < 256; k0 += 32) {
      v16h af;
      const half_t* ap = Ab + row * 264 + k0 + hi * 8;
      ((float4*)&af)[0] = *(const float4*)ap;
      ((float4*)&af)[1] = *(const float4*)(ap + 16);
      #pragma unroll
      for (int t = 0; t < 2; ++t) {
        int j = jbase + t * 16;
        v16h bf = load16(fpw_t + (size_t)j * 2048 + kc * 256 + k0 + hi * 8);
        acc[t] = __builtin_amdgcn_wmma_f32_16x16x32_f16(false, af, false, bf,
                                                        (short)0, acc[t], false, false);
      }
    }
  }
  #pragma unroll
  for (int t = 0; t < 2; ++t) {
    int j = jbase + t * 16;
    float bias = fp_b[j];
    #pragma unroll
    for (int i = 0; i < 8; ++i) {
      int M = i + hi * 8;
      size_t ridx = (size_t)(b * NN + n0 + M) * DIM + j;
      float v = acc[t][i] + bias + feats[ridx];
      out[ridx] = 1.0f / (1.0f + __expf(-v));
    }
  }
}

extern "C" void kernel_launch(void* const* d_in, const int* in_sizes, int n_in,
                              void* d_out, int out_size, void* d_ws, size_t ws_size,
                              hipStream_t stream) {
  (void)in_sizes; (void)n_in; (void)out_size; (void)ws_size;
  const float* feats = (const float*)d_in[0];
  const float* adj   = (const float*)d_in[1];
  const float* fc_w  = (const float*)d_in[2];
  const float* fc_b  = (const float*)d_in[3];
  const float* q_w   = (const float*)d_in[4];
  const float* q_b   = (const float*)d_in[5];
  const float* k_w   = (const float*)d_in[6];
  const float* k_b   = (const float*)d_in[7];
  const float* fp_w  = (const float*)d_in[8];
  const float* fp_b  = (const float*)d_in[9];
  float* out = (float*)d_out;
  char* ws = (char*)d_ws;

  half_t* feats_h = (half_t*)(ws + OFF_FEATS_H);
  half_t* fcw_t   = (half_t*)(ws + OFF_FCW_T);
  half_t* fpw_t   = (half_t*)(ws + OFF_FPW_T);
  half_t* fpT     = (half_t*)(ws + OFF_FPT);
  float*  Q       = (float*)(ws + OFF_Q);
  float*  Kv      = (float*)(ws + OFF_K);
  half_t* outh    = (half_t*)(ws + OFF_OUTH);

  k_convert<<<12288, 256, 0, stream>>>(feats, fc_w, fp_w, feats_h, fcw_t, fpw_t);
  k_proj  <<<BS * 64 * HEADS, 256, 0, stream>>>(feats_h, fcw_t, fc_b, fpT);
  k_qk    <<<BS * HEADS * 4, 256, 0, stream>>>(fpT, q_w, q_b, k_w, k_b, Q, Kv);
  k_attn  <<<BS * 64, 256, 0, stream>>>(adj, fpT, Q, Kv, outh);
  k_final <<<BS * 64, 256, 0, stream>>>(outh, fpw_t, fp_b, feats, out);
}
